// MyRNN_57088705298741
// MI455X (gfx1250) — matrix operand order repaired and checked by reading, not compiled, not run on hardware
//
#include <hip/hip_runtime.h>
#include <hip/hip_bf16.h>

// ---- problem dims ----
#define T_LEN 2048
#define B_SZ  64
#define IN_F  256
#define HID   512
#define OUT_F 256

// ---- tiling ----
#define M_PER_WG 16
#define N_WG (B_SZ / M_PER_WG)      // 4 workgroups (batch split)
#define WAVES 16
#define THREADS (WAVES * 32)        // 512 threads, wave32

#define KT_H 16                     // K tiles over hidden (512/32)
#define KT_X 8                      // K tiles over input  (256/32)
#define KT_W1 (KT_H + KT_X)         // 24 K tiles for [h;x] @ W1
#define NT_H 32                     // N tiles over hidden (512/16)
#define NT_Y 16                     // N tiles over output (256/16)
#define FRAG_ELEMS 512              // 32 lanes * 16 bf16 per fragment

#define W1F_ELEMS (NT_H * KT_W1 * FRAG_ELEMS)               // 393216 bf16
#define W2F_ELEMS (NT_Y * KT_H * FRAG_ELEMS)                // 131072 bf16
#define HSEQ_ELEMS ((size_t)T_LEN * N_WG * KT_H * FRAG_ELEMS) // 64M bf16 = 128MB

typedef __attribute__((ext_vector_type(16))) __bf16 v16bf;
typedef __attribute__((ext_vector_type(8)))  float  v8f;

#define WMMA_BF16(a, b, c) \
    __builtin_amdgcn_wmma_f32_16x16x32_bf16(false, (a), false, (b), (short)0, (c), false, false)

// 16-bit A fragment (16x32) element->K mapping (ISA 7.12.2):
// lanes 0-15 : e0..7 -> K0..7,  e8..15 -> K16..23
// lanes 16-31: e0..7 -> K8..15, e8..15 -> K24..31
__device__ __forceinline__ int a_kk(int slot_lane, int e) {
    int kk = (e < 8) ? e : (e + 8);
    if (slot_lane >= 16) kk += 8;
    return kk;
}
// inverse: (m, kk) -> slot index (lane*16 + e) within a fragment
__device__ __forceinline__ int a_slot(int m, int kk) {
    int sl = m + (((kk >> 3) & 1) ? 16 : 0);
    int e  = (kk & 7) + ((kk >= 16) ? 8 : 0);
    return sl * 16 + e;
}

// Pre-swizzle W1 (768x512) and W2 (512x256) f32 -> bf16 B-fragment layout.
// B fragment (32x16): lane l holds column N = l&15, elem e holds K = a_kk(l, e) (same K order as A).
__global__ void __launch_bounds__(512) wprep_kernel(
    const float* __restrict__ W1, const float* __restrict__ W2,
    __bf16* __restrict__ ws)
{
    int f   = blockIdx.x;
    int tid = threadIdx.x;
    int slot_lane = tid >> 4;
    int e  = tid & 15;
    int nn = slot_lane & 15;
    int kk = a_kk(slot_lane, e);

    if (f < NT_H * KT_W1) {                       // W1 fragments: [nt][kt]
        int nt = f / KT_W1, kt = f % KT_W1;
        int row = (kt < KT_H) ? (kt * 32 + kk)
                              : (HID + (kt - KT_H) * 32 + kk);
        int col = nt * 16 + nn;
        ws[(size_t)f * FRAG_ELEMS + tid] = (__bf16)W1[(size_t)row * HID + col];
    } else {                                      // W2 fragments: [nt][kt]
        int g  = f - NT_H * KT_W1;
        int nt = g / KT_H, kt = g % KT_H;
        int row = kt * 32 + kk;
        int col = nt * 16 + nn;
        ws[(size_t)W1F_ELEMS + (size_t)g * FRAG_ELEMS + tid] =
            (__bf16)W2[(size_t)row * OUT_F + col];
    }
}

template<bool DEFER>
__global__ void __launch_bounds__(THREADS) rnn_scan_kernel(
    const float* __restrict__ x, const float* __restrict__ b1,
    const float* __restrict__ b2, const __bf16* __restrict__ ws,
    __bf16* __restrict__ hseq, float* __restrict__ out)
{
    // H double buffer + double-buffered x staging, all in A-fragment layout (bf16)
    __shared__ __align__(32) __bf16 Hbuf[2][KT_H * FRAG_ELEMS];  // 2 x 16 KB
    __shared__ __align__(32) __bf16 Xbuf[2][KT_X * FRAG_ELEMS];  // 2 x 8 KB

    const int tid  = threadIdx.x;
    const int w    = tid >> 5;          // wave id 0..15
    const int lane = tid & 31;
    const int mb   = blockIdx.x * M_PER_WG;

    const __bf16* __restrict__ W1f = ws;
    const __bf16* __restrict__ W2f = ws + W1F_ELEMS;

    // h0 = 0
    for (int i = tid; i < KT_H * FRAG_ELEMS; i += THREADS)
        Hbuf[0][i] = (__bf16)0.0f;

    // per-wave tile ownership: hidden n-tiles {2w,2w+1}, output n-tile w
    const int nt0 = 2 * w, nt1 = 2 * w + 1;
    const float b1v0 = b1[nt0 * 16 + (lane & 15)];
    const float b1v1 = b1[nt1 * 16 + (lane & 15)];
    const float b2v  = b2[w * 16 + (lane & 15)];
    const int mrow = (lane >= 16) ? 8 : 0;      // C/D layout: upper lanes -> rows 8..15

    // fragment-slot decode for staging/store done by this thread (slot == tid)
    const int s_lane = tid >> 4, s_e = tid & 15;
    const int s_m  = s_lane & 15;
    const int s_kk = a_kk(s_lane, s_e);
    const float* xbase = x + (size_t)(mb + s_m) * ((size_t)T_LEN * IN_F);

    // stage x_0
    #pragma unroll
    for (int kt = 0; kt < KT_X; ++kt)
        Xbuf[0][kt * FRAG_ELEMS + tid] = (__bf16)xbase[kt * 32 + s_kk];

    int cur = 0;
    __syncthreads();

    for (int t = 0; t < T_LEN; ++t) {
        const int xp = t & 1;

        // ---- early-issue global loads of x_{t+1}: drain under the recurrent GEMM ----
        float xv[KT_X];
        const bool more = (t + 1 < T_LEN);
        if (more) {
            const float* xr = xbase + (size_t)(t + 1) * IN_F;
            __builtin_prefetch(xr + IN_F, 0, 0);   // global_prefetch_b8: warm L2 for t+2
            #pragma unroll
            for (int kt = 0; kt < KT_X; ++kt) xv[kt] = xr[kt * 32 + s_kk];
        }

        // ---- recurrent GEMM: [h;x](16x768) @ W1(768x512); 4 chains, A preloaded ----
        const __bf16* hsrc = &Hbuf[cur][0];
        const __bf16* xsrc = &Xbuf[xp][0];
        v8f acc0a = {}, acc0b = {}, acc1a = {}, acc1b = {};
        v16bf a0 = *(const v16bf*)(hsrc + 0 * FRAG_ELEMS + lane * 16);
        v16bf a1 = *(const v16bf*)(hsrc + 1 * FRAG_ELEMS + lane * 16);
        #pragma unroll
        for (int kt = 0; kt < KT_W1; kt += 2) {
            v16bf a2 = a0, a3 = a1;
            if (kt + 2 < KT_W1) {
                a2 = (kt + 2 < KT_H)
                   ? *(const v16bf*)(hsrc + (kt + 2) * FRAG_ELEMS + lane * 16)
                   : *(const v16bf*)(xsrc + (kt + 2 - KT_H) * FRAG_ELEMS + lane * 16);
                a3 = (kt + 3 < KT_H)
                   ? *(const v16bf*)(hsrc + (kt + 3) * FRAG_ELEMS + lane * 16)
                   : *(const v16bf*)(xsrc + (kt + 3 - KT_H) * FRAG_ELEMS + lane * 16);
            }
            v16bf b00 = *(const v16bf*)(W1f + ((size_t)(nt0 * KT_W1 + kt    )) * FRAG_ELEMS + lane * 16);
            v16bf b01 = *(const v16bf*)(W1f + ((size_t)(nt0 * KT_W1 + kt + 1)) * FRAG_ELEMS + lane * 16);
            v16bf b10 = *(const v16bf*)(W1f + ((size_t)(nt1 * KT_W1 + kt    )) * FRAG_ELEMS + lane * 16);
            v16bf b11 = *(const v16bf*)(W1f + ((size_t)(nt1 * KT_W1 + kt + 1)) * FRAG_ELEMS + lane * 16);
            acc0a = WMMA_BF16(a0, b00, acc0a);
            acc1a = WMMA_BF16(a0, b10, acc1a);
            acc0b = WMMA_BF16(a1, b01, acc0b);
            acc1b = WMMA_BF16(a1, b11, acc1b);
            a0 = a2; a1 = a3;
        }

        // ---- tanh + scatter h_new into next A buffer (this wave owns k-tile w) ----
        const int nxt = cur ^ 1;
        __bf16* hdst = &Hbuf[nxt][w * FRAG_ELEMS];
        #pragma unroll
        for (int v = 0; v < 8; ++v) {
            int m = mrow + v;
            float h0 = tanhf(acc0a[v] + acc0b[v] + b1v0);  // n = 32w    + lane%16
            float h1 = tanhf(acc1a[v] + acc1b[v] + b1v1);  // n = 32w+16 + lane%16
            hdst[a_slot(m, (lane & 15))]      = (__bf16)h0;
            hdst[a_slot(m, 16 + (lane & 15))] = (__bf16)h1;
        }

        if (DEFER) {
            // publish h_t fragment to global Hseq: this wave wrote all of k-tile w
            // itself (same-wave DS ordering), so no barrier needed before readback.
            v16bf hv = *(const v16bf*)(&Hbuf[nxt][w * FRAG_ELEMS + lane * 16]);
            *(v16bf*)(hseq + (((size_t)t * N_WG + blockIdx.x) * KT_H + w) * FRAG_ELEMS
                      + lane * 16) = hv;
        } else {
            __syncthreads();   // publish full H_new for the fused output GEMM
            v8f ya = {}, yb = {};
            const __bf16* hn = &Hbuf[nxt][0];
            v16bf c0 = *(const v16bf*)(hn + 0 * FRAG_ELEMS + lane * 16);
            v16bf c1 = *(const v16bf*)(hn + 1 * FRAG_ELEMS + lane * 16);
            #pragma unroll
            for (int kt = 0; kt < KT_H; kt += 2) {
                v16bf c2 = c0, c3 = c1;
                if (kt + 2 < KT_H) {
                    c2 = *(const v16bf*)(hn + (kt + 2) * FRAG_ELEMS + lane * 16);
                    c3 = *(const v16bf*)(hn + (kt + 3) * FRAG_ELEMS + lane * 16);
                }
                v16bf bb0 = *(const v16bf*)(W2f + ((size_t)(w * KT_H + kt    )) * FRAG_ELEMS + lane * 16);
                v16bf bb1 = *(const v16bf*)(W2f + ((size_t)(w * KT_H + kt + 1)) * FRAG_ELEMS + lane * 16);
                ya = WMMA_BF16(c0, bb0, ya);
                yb = WMMA_BF16(c1, bb1, yb);
                c0 = c2; c1 = c3;
            }
            #pragma unroll
            for (int v = 0; v < 8; ++v) {
                int m = mrow + v;
                out[(size_t)(mb + m) * ((size_t)T_LEN * OUT_F)
                    + (size_t)t * OUT_F + w * 16 + (lane & 15)] = ya[v] + yb[v] + b2v;
            }
        }

        // ---- publish x_{t+1} staging (loads issued at top of this iteration) ----
        if (more) {
            #pragma unroll
            for (int kt = 0; kt < KT_X; ++kt)
                Xbuf[xp ^ 1][kt * FRAG_ELEMS + tid] = (__bf16)xv[kt];
        }
        __syncthreads();
        cur = nxt;
    }

    // ---- h_last (f32) appended after the y output block ----
    float* hlast = out + (size_t)B_SZ * T_LEN * OUT_F;
    for (int kt = 0; kt < KT_H; ++kt)
        hlast[(size_t)(mb + s_m) * HID + kt * 32 + s_kk] =
            (float)Hbuf[cur][kt * FRAG_ELEMS + tid];
}

// Fully parallel deferred output projection: one WG per (t, batch-tile),
// y = h_t @ W2 + b2 from fragment-layout Hseq.
__global__ void __launch_bounds__(THREADS) yproj_kernel(
    const __bf16* __restrict__ ws, const __bf16* __restrict__ hseq,
    const float* __restrict__ b2, float* __restrict__ out)
{
    __shared__ __align__(32) __bf16 Ablk[KT_H * FRAG_ELEMS];   // 16 KB
    const int tid  = threadIdx.x;
    const int w    = tid >> 5;
    const int lane = tid & 31;
    const int t    = blockIdx.x >> 2;
    const int wgm  = blockIdx.x & (N_WG - 1);
    const __bf16* __restrict__ W2f = ws + W1F_ELEMS;
    const __bf16* __restrict__ src = hseq + (size_t)blockIdx.x * (KT_H * FRAG_ELEMS);

    // cooperative stage of the 16 A fragments (32 B per thread, coalesced)
    *(v16bf*)(&Ablk[tid * 16]) = *(const v16bf*)(src + tid * 16);
    __syncthreads();

    v8f ya = {}, yb = {};
    v16bf a0 = *(const v16bf*)(&Ablk[0 * FRAG_ELEMS + lane * 16]);
    v16bf a1 = *(const v16bf*)(&Ablk[1 * FRAG_ELEMS + lane * 16]);
    #pragma unroll
    for (int kt = 0; kt < KT_H; kt += 2) {
        v16bf a2 = a0, a3 = a1;
        if (kt + 2 < KT_H) {
            a2 = *(const v16bf*)(&Ablk[(kt + 2) * FRAG_ELEMS + lane * 16]);
            a3 = *(const v16bf*)(&Ablk[(kt + 3) * FRAG_ELEMS + lane * 16]);
        }
        v16bf bb0 = *(const v16bf*)(W2f + ((size_t)(w * KT_H + kt    )) * FRAG_ELEMS + lane * 16);
        v16bf bb1 = *(const v16bf*)(W2f + ((size_t)(w * KT_H + kt + 1)) * FRAG_ELEMS + lane * 16);
        ya = WMMA_BF16(a0, bb0, ya);
        yb = WMMA_BF16(a1, bb1, yb);
        a0 = a2; a1 = a3;
    }

    const float b2v = b2[w * 16 + (lane & 15)];
    const int mrow = (lane >= 16) ? 8 : 0;
    #pragma unroll
    for (int v = 0; v < 8; ++v) {
        int m = mrow + v;
        out[(size_t)(wgm * 16 + m) * ((size_t)T_LEN * OUT_F)
            + (size_t)t * OUT_F + w * 16 + (lane & 15)] = ya[v] + yb[v] + b2v;
    }
}

extern "C" void kernel_launch(void* const* d_in, const int* in_sizes, int n_in,
                              void* d_out, int out_size, void* d_ws, size_t ws_size,
                              hipStream_t stream) {
    const float* x  = (const float*)d_in[0];   // (64, 2048, 256)
    const float* W1 = (const float*)d_in[1];   // (768, 512)
    const float* b1 = (const float*)d_in[2];   // (512,)
    const float* W2 = (const float*)d_in[3];   // (512, 256)
    const float* b2 = (const float*)d_in[4];   // (256,)
    float*  out = (float*)d_out;               // y (64,2048,256) ++ h_last (1,64,512)
    __bf16* ws  = (__bf16*)d_ws;

    wprep_kernel<<<NT_H * KT_W1 + NT_Y * KT_H, 512, 0, stream>>>(W1, W2, ws);

    const size_t need = ((size_t)W1F_ELEMS + W2F_ELEMS + HSEQ_ELEMS) * sizeof(__bf16);
    __bf16* hseq = ws + W1F_ELEMS + W2F_ELEMS;
    if (ws_size >= need) {
        // scan critical path carries only the recurrent GEMM; projection runs
        // afterwards across the whole chip.
        rnn_scan_kernel<true><<<N_WG, THREADS, 0, stream>>>(x, b1, b2, ws, hseq, out);
        yproj_kernel<<<T_LEN * N_WG, THREADS, 0, stream>>>(ws, hseq, b2, out);
    } else {
        rnn_scan_kernel<false><<<N_WG, THREADS, 0, stream>>>(x, b1, b2, ws, ws, out);
    }
}
